// SelfAttention_68693706932535
// MI455X (gfx1250) — matrix operand order repaired and checked by reading, not compiled, not run on hardware
//
#include <hip/hip_runtime.h>
#include <math.h>

// ---------------------------------------------------------------------------
// SelfAttention forward for MI455X (gfx1250, wave32, WMMA).
// Heavy einsums -> v_wmma_f32_16x16x32_bf16 (f32 accumulation).
// Global->LDS staging -> global_load_async_to_lds_b128 (ASYNCcnt-tracked DMA),
// removing the VGPR relay that spilled to scratch in the previous round.
// Attention is flash-style (no SxS score matrix in HBM).
// ---------------------------------------------------------------------------

typedef __attribute__((ext_vector_type(16))) __bf16 v16bf;
typedef __attribute__((ext_vector_type(8)))  float  v8f;

#define WMMA_BF16(a, b, c) \
  __builtin_amdgcn_wmma_f32_16x16x32_bf16(false, (a), false, (b), (short)0, (c), false, false)

// ---- CDNA5 async global->LDS DMA (16B per lane), tracked with ASYNCcnt -----
__device__ inline void async_copy16(void* lds, const void* gptr) {
  const unsigned lds_addr = (unsigned)(unsigned long long)lds;  // flat->LDS: addr[31:0]
  const unsigned long long ga = (unsigned long long)gptr;
  asm volatile("global_load_async_to_lds_b128 %0, %1, off"
               :
               : "v"(lds_addr), "v"(ga)
               : "memory");
}
__device__ inline void wait_async() {
  asm volatile("s_wait_asynccnt 0x0" ::: "memory");
}

// ---------------- WMMA fragment loaders (contiguous k = 16*half + s in A and B) ----------

// A fragment: 16(M) x 32(K) bf16 tile, row-major with leading dim `ld`.
__device__ inline v16bf load_frag_a(const __bf16* p, int ld) {
  const int lane = threadIdx.x & 31;
  const int half = lane >> 4;
  const int r    = lane & 15;
  v16bf a;
#pragma unroll
  for (int j = 0; j < 8; ++j) {
    const int k = half * 16 + 2 * j;
    a[2 * j]     = p[r * ld + k];
    a[2 * j + 1] = p[r * ld + k + 1];
  }
  return a;
}

// B fragment from a tile stored as [N rows][K cols] (W row-major, B = W^T).
__device__ inline v16bf load_frag_b_nk(const __bf16* p, int ld) {
  const int lane = threadIdx.x & 31;
  const int half = lane >> 4;
  const int n    = lane & 15;
  v16bf b;
#pragma unroll
  for (int j = 0; j < 8; ++j) {
    const int k = half * 16 + 2 * j;
    b[2 * j]     = p[n * ld + k];
    b[2 * j + 1] = p[n * ld + k + 1];
  }
  return b;
}

// B fragment from a tile stored as [K rows][N cols] (e.g. V chunk).
__device__ inline v16bf load_frag_b_kn(const __bf16* p, int ld) {
  const int lane = threadIdx.x & 31;
  const int half = lane >> 4;
  const int n    = lane & 15;
  v16bf b;
#pragma unroll
  for (int j = 0; j < 8; ++j) {
    const int k = half * 16 + 2 * j;
    b[2 * j]     = p[k * ld + n];
    b[2 * j + 1] = p[(k + 1) * ld + n];
  }
  return b;
}

// 16-lane-group reductions (wave32: halves hold distinct rows)
__device__ inline float hmax16(float v) {
  v = fmaxf(v, __shfl_xor(v, 1, 16));
  v = fmaxf(v, __shfl_xor(v, 2, 16));
  v = fmaxf(v, __shfl_xor(v, 4, 16));
  v = fmaxf(v, __shfl_xor(v, 8, 16));
  return v;
}
__device__ inline float hsum16(float v) {
  v += __shfl_xor(v, 1, 16);
  v += __shfl_xor(v, 2, 16);
  v += __shfl_xor(v, 4, 16);
  v += __shfl_xor(v, 8, 16);
  return v;
}

// ---------------- fp32 -> bf16 conversion -----------------------------------

__global__ __launch_bounds__(256) void f32_to_bf16_k(const float* __restrict__ s,
                                                     __bf16* __restrict__ d) {
  const size_t i = (size_t)blockIdx.x * 256 + threadIdx.x;
  d[i] = (__bf16)s[i];
}

// ---------------- Generic bf16 GEMM: C[M,N=1024] = A[M,K] * W[N,K]^T --------
// mode 0: + bias[n] + posenc[(m%2048), n]; write Cf (f32) and Cb (bf16)
// mode 1: + bias[n];                       write Cb (bf16)
// mode 2: tanh(.);                         write Cf (f32)
// 128x128 C tile, 8 waves (4x2), K staged in 32-wide chunks, double-buffered
// LDS fed by async global->LDS DMA (no VGPR staging, no spills).
__global__ __launch_bounds__(256) void gemm_bf16(
    const __bf16* __restrict__ A, const __bf16* __restrict__ W,
    const float* __restrict__ bias, const float* __restrict__ posenc,
    float* __restrict__ Cf, __bf16* __restrict__ Cb, int K, int mode) {
  constexpr int TM = 128, TN = 128, KC = 32;
  __shared__ __bf16 sA[2][TM * KC];
  __shared__ __bf16 sW[2][TN * KC];

  const int tid  = threadIdx.x;
  const int m0   = blockIdx.y * TM;
  const int n0   = blockIdx.x * TN;
  const int wave = tid >> 5;
  const int lane = tid & 31;
  const int half = lane >> 4;
  const int col  = lane & 15;
  const int wm   = (wave >> 1) * 32;
  const int wn   = (wave & 1) * 64;

  auto stage = [&](int kc, int buf) {
#pragma unroll
    for (int l = 0; l < 2; ++l) {
      const int idx = tid + l * 256;   // 16B unit within the 128x32 tile
      const int row = idx >> 2;
      const int c8  = (idx & 3) << 3;
      async_copy16(&sA[buf][row * KC + c8], A + (size_t)(m0 + row) * K + kc + c8);
      async_copy16(&sW[buf][row * KC + c8], W + (size_t)(n0 + row) * K + kc + c8);
    }
  };

  v8f acc[2][4];
#pragma unroll
  for (int i = 0; i < 2; ++i)
#pragma unroll
    for (int j = 0; j < 4; ++j)
#pragma unroll
      for (int e = 0; e < 8; ++e) acc[i][j][e] = 0.f;

  const int nk = K / KC;
  stage(0, 0);
  wait_async();
  __syncthreads();

  int buf = 0;
  for (int ks = 0; ks < nk; ++ks) {
    if (ks + 1 < nk) stage((ks + 1) * KC, buf ^ 1);
    v16bf af[2], bfz[4];
#pragma unroll
    for (int i = 0; i < 2; ++i) af[i] = load_frag_a(&sA[buf][(wm + i * 16) * KC], KC);
#pragma unroll
    for (int j = 0; j < 4; ++j) bfz[j] = load_frag_b_nk(&sW[buf][(wn + j * 16) * KC], KC);
#pragma unroll
    for (int i = 0; i < 2; ++i)
#pragma unroll
      for (int j = 0; j < 4; ++j) acc[i][j] = WMMA_BF16(af[i], bfz[j], acc[i][j]);
    wait_async();
    __syncthreads();
    buf ^= 1;
  }

  // epilogue: c[e] at (m = e + 8*half, n = lane&15) inside each 16x16 subtile
#pragma unroll
  for (int i = 0; i < 2; ++i) {
#pragma unroll
    for (int j = 0; j < 4; ++j) {
#pragma unroll
      for (int e = 0; e < 8; ++e) {
        const int gm = m0 + wm + i * 16 + half * 8 + e;
        const int gn = n0 + wn + j * 16 + col;
        const size_t idx = (size_t)gm * 1024 + gn;
        float v = acc[i][j][e];
        if (mode == 0) {
          v += bias[gn] + posenc[(size_t)(gm & 2047) * 1024 + gn];
          Cf[idx] = v;
          Cb[idx] = (__bf16)v;
        } else if (mode == 1) {
          v += bias[gn];
          Cb[idx] = (__bf16)v;
        } else {
          Cf[idx] = tanhf(v);
        }
      }
    }
  }
}

// ---------------- Flash attention ------------------------------------------
// One workgroup = 16 query rows of one batch. 8 waves; wave w owns S-subtile
// columns [16w,16w+16) of the 128-wide key block and output dims [128w,128w+128).
// K chunks double-buffered via async DMA; online softmax state in LDS.
__global__ __launch_bounds__(256) void flash_attn(
    const __bf16* __restrict__ Qm, const __bf16* __restrict__ Km,
    const __bf16* __restrict__ Vm, float* __restrict__ Zf) {
  constexpr int Dd = 1024, S = 2048;
  extern __shared__ char smem[];
  __bf16* sQ = (__bf16*)smem;       // 16 x 1024        (32 KB)
  __bf16* sK = sQ + 16 * 1024;      // 2 x (128 x 32)   (16 KB)
  __bf16* sP = sK + 2 * 128 * 32;   // 16 x 128         ( 4 KB)
  __bf16* sV = sP + 16 * 128;       // 32 x 1024        (64 KB)
  float*  fr = (float*)(sV + 32 * 1024);
  float* rm  = fr;        // [8][16] per-wave row maxima
  float* rs  = fr + 128;  // [8][16] per-wave row sums
  float* stm = fr + 256;  // [16] running max
  float* stl = fr + 272;  // [16] running denom
  float* nmx = fr + 288;  // [16] new max
  float* scl = fr + 304;  // [16] rescale factor

  const int tid  = threadIdx.x;
  const int wave = tid >> 5;
  const int lane = tid & 31;
  const int half = lane >> 4;
  const int col  = lane & 15;
  const int b    = blockIdx.y;
  const int rb   = blockIdx.x;
  const size_t qbase = ((size_t)b * S + rb * 16) * Dd;

  // stage the 16-row Q block (contiguous) via async DMA
#pragma unroll
  for (int l = 0; l < 8; ++l) {
    const int idx = tid + l * 256;
    async_copy16(sQ + idx * 8, Qm + qbase + (size_t)idx * 8);
  }
  if (tid < 16) { stm[tid] = -INFINITY; stl[tid] = 0.f; }

  auto stageK = [&](int j0, int kc, int kb) {
#pragma unroll
    for (int l = 0; l < 2; ++l) {
      const int idx = tid + l * 256;
      const int row = idx >> 2;
      const int c8  = (idx & 3) << 3;
      async_copy16(sK + kb * (128 * 32) + row * 32 + c8,
                   Km + ((size_t)b * S + j0 + row) * Dd + kc + c8);
    }
  };
  auto stageV = [&](int j0, int kc2) {
#pragma unroll
    for (int l = 0; l < 16; ++l) {
      const int idx = tid + l * 256;
      async_copy16(sV + idx * 8, Vm + ((size_t)b * S + j0 + kc2) * Dd + (size_t)idx * 8);
    }
  };

  v8f o[8];
#pragma unroll
  for (int nt = 0; nt < 8; ++nt)
#pragma unroll
    for (int e = 0; e < 8; ++e) o[nt][e] = 0.f;

  for (int jb = 0; jb < S / 128; ++jb) {
    const int j0 = jb * 128;

    // ---- S subtile (16x16 per wave), K=1024 reduction, double-buffered ----
    v8f sacc;
#pragma unroll
    for (int e = 0; e < 8; ++e) sacc[e] = 0.f;

    stageK(j0, 0, 0);
    wait_async();        // also covers the Q staging on jb==0
    __syncthreads();
    int kb = 0;
    for (int kc = 0; kc < Dd; kc += 32) {
      if (kc + 32 < Dd) stageK(j0, kc + 32, kb ^ 1);
      v16bf a  = load_frag_a(sQ + kc, Dd);
      v16bf bb = load_frag_b_nk(sK + kb * (128 * 32) + wave * 16 * 32, 32);
      sacc = WMMA_BF16(a, bb, sacc);
      wait_async();
      __syncthreads();
      kb ^= 1;
    }
#pragma unroll
    for (int e = 0; e < 8; ++e) sacc[e] *= 0.03125f;  // 1/sqrt(1024)

    // ---- cross-wave online softmax ----
#pragma unroll
    for (int e = 0; e < 8; ++e) {
      const float v = hmax16(sacc[e]);
      if (col == 0) rm[wave * 16 + half * 8 + e] = v;
    }
    __syncthreads();
    if (tid < 16) {
      float mx = stm[tid];
#pragma unroll
      for (int w2 = 0; w2 < 8; ++w2) mx = fmaxf(mx, rm[w2 * 16 + tid]);
      scl[tid] = __expf(stm[tid] - mx);
      nmx[tid] = mx;
      stm[tid] = mx;
    }
    __syncthreads();

#pragma unroll
    for (int nt = 0; nt < 8; ++nt)
#pragma unroll
      for (int e = 0; e < 8; ++e) o[nt][e] *= scl[half * 8 + e];

#pragma unroll
    for (int e = 0; e < 8; ++e) {
      const int r = half * 8 + e;
      const float pv = __expf(sacc[e] - nmx[r]);
      sacc[e] = pv;
      sP[r * 128 + wave * 16 + col] = (__bf16)pv;
    }
#pragma unroll
    for (int e = 0; e < 8; ++e) {
      const float v = hsum16(sacc[e]);
      if (col == 0) rs[wave * 16 + half * 8 + e] = v;
    }
    __syncthreads();
    if (tid < 16) {
      float s = 0.f;
#pragma unroll
      for (int w2 = 0; w2 < 8; ++w2) s += rs[w2 * 16 + tid];
      stl[tid] = stl[tid] * scl[tid] + s;
    }

    // ---- O += P(16x128) x V(128 x 128_w), 32-row async-staged chunks ----
    for (int kc2 = 0; kc2 < 128; kc2 += 32) {
      stageV(j0, kc2);
      wait_async();
      __syncthreads();
      v16bf a = load_frag_a(sP + kc2, 128);
#pragma unroll
      for (int nt = 0; nt < 8; ++nt) {
        v16bf bb = load_frag_b_kn(sV + wave * 128 + nt * 16, Dd);
        o[nt] = WMMA_BF16(a, bb, o[nt]);
      }
      __syncthreads();
    }
  }

  // ---- epilogue: Z = O / l ----
#pragma unroll
  for (int nt = 0; nt < 8; ++nt) {
#pragma unroll
    for (int e = 0; e < 8; ++e) {
      const int r = half * 8 + e;
      const float z = o[nt][e] / stl[r];
      Zf[((size_t)b * S + rb * 16 + r) * Dd + wave * 128 + nt * 16 + col] = z;
    }
  }
}

// ---------------- add + LayerNorm ------------------------------------------
__global__ __launch_bounds__(256) void add_layernorm(
    const float* __restrict__ xf, const float* __restrict__ Zf,
    const float* __restrict__ g, const float* __restrict__ bta,
    float* __restrict__ Hf, __bf16* __restrict__ Hb) {
  __shared__ float red[256];
  const int tid = threadIdx.x;
  const size_t base = (size_t)blockIdx.x * 1024;
  float v[4];
  float s = 0.f;
#pragma unroll
  for (int i = 0; i < 4; ++i) {
    const int c = tid + i * 256;
    v[i] = xf[base + c] + Zf[base + c];
    s += v[i];
  }
  red[tid] = s;
  __syncthreads();
  for (int o2 = 128; o2 > 0; o2 >>= 1) {
    if (tid < o2) red[tid] += red[tid + o2];
    __syncthreads();
  }
  const float mu = red[0] * (1.f / 1024.f);
  __syncthreads();
  float s2 = 0.f;
#pragma unroll
  for (int i = 0; i < 4; ++i) {
    const float d = v[i] - mu;
    s2 += d * d;
  }
  red[tid] = s2;
  __syncthreads();
  for (int o2 = 128; o2 > 0; o2 >>= 1) {
    if (tid < o2) red[tid] += red[tid + o2];
    __syncthreads();
  }
  const float rstd = rsqrtf(red[0] * (1.f / 1024.f) + 1e-5f);
#pragma unroll
  for (int i = 0; i < 4; ++i) {
    const int c = tid + i * 256;
    const float h = (v[i] - mu) * rstd * g[c] + bta[c];
    Hf[base + c] = h;
    Hb[base + c] = (__bf16)h;
  }
}

// ---------------- soft_key . soft_query -------------------------------------
__global__ __launch_bounds__(256) void softkey_dot(const float* __restrict__ SK,
                                                   const float* __restrict__ q,
                                                   float* __restrict__ t) {
  __shared__ float red[256];
  const int tid = threadIdx.x;
  const size_t base = (size_t)blockIdx.x * 1024;
  float s = 0.f;
#pragma unroll
  for (int i = 0; i < 4; ++i) {
    const int c = tid + i * 256;
    s += SK[base + c] * q[c];
  }
  red[tid] = s;
  __syncthreads();
  for (int o2 = 128; o2 > 0; o2 >>= 1) {
    if (tid < o2) red[tid] += red[tid + o2];
    __syncthreads();
  }
  if (tid == 0) t[blockIdx.x] = red[0];
}

// ---------------- softmax over sequence (per batch) -------------------------
__global__ __launch_bounds__(256) void seq_softmax(const float* __restrict__ t,
                                                   float* __restrict__ p) {
  __shared__ float red[256];
  const int tid = threadIdx.x;
  const size_t base = (size_t)blockIdx.x * 2048;
  float v[8];
  float mx = -INFINITY;
#pragma unroll
  for (int i = 0; i < 8; ++i) {
    v[i] = t[base + tid + i * 256];
    mx = fmaxf(mx, v[i]);
  }
  red[tid] = mx;
  __syncthreads();
  for (int o2 = 128; o2 > 0; o2 >>= 1) {
    if (tid < o2) red[tid] = fmaxf(red[tid], red[tid + o2]);
    __syncthreads();
  }
  mx = red[0];
  __syncthreads();
  float s = 0.f;
#pragma unroll
  for (int i = 0; i < 8; ++i) {
    v[i] = __expf(v[i] - mx);
    s += v[i];
  }
  red[tid] = s;
  __syncthreads();
  for (int o2 = 128; o2 > 0; o2 >>= 1) {
    if (tid < o2) red[tid] += red[tid + o2];
    __syncthreads();
  }
  const float inv = 1.f / red[0];
#pragma unroll
  for (int i = 0; i < 8; ++i) p[base + tid + i * 256] = v[i] * inv;
}

// ---------------- out[b,o] = sum_s p[b,s] * H[b,s,o] ------------------------
__global__ __launch_bounds__(256) void weighted_sum(const float* __restrict__ p,
                                                    const float* __restrict__ H,
                                                    float* __restrict__ out) {
  const int o = blockIdx.x * 256 + threadIdx.x;
  const int b = blockIdx.y;
  float acc = 0.f;
  for (int s = 0; s < 2048; ++s)
    acc += p[(size_t)b * 2048 + s] * H[((size_t)b * 2048 + s) * 1024 + o];
  out[(size_t)b * 1024 + o] = acc;
}

// ---------------------------------------------------------------------------
extern "C" void kernel_launch(void* const* d_in, const int* in_sizes, int n_in,
                              void* d_out, int out_size, void* d_ws, size_t ws_size,
                              hipStream_t stream) {
  (void)in_sizes; (void)n_in; (void)out_size; (void)ws_size;
  constexpr int D_IN = 512, D = 1024, SEQ = 2048, B = 8, M = B * SEQ;

  const float* inputs  = (const float*)d_in[0];
  const float* embed_w = (const float*)d_in[1];
  const float* embed_b = (const float*)d_in[2];
  const float* wq_w = (const float*)d_in[3];
  const float* wq_b = (const float*)d_in[4];
  const float* wk_w = (const float*)d_in[5];
  const float* wk_b = (const float*)d_in[6];
  const float* wv_w = (const float*)d_in[7];
  const float* wv_b = (const float*)d_in[8];
  const float* ln_g = (const float*)d_in[9];
  const float* ln_b = (const float*)d_in[10];
  const float* sw_w = (const float*)d_in[11];
  const float* sq   = (const float*)d_in[12];
  const float* pos  = (const float*)d_in[13];
  float* out = (float*)d_out;

  char* w = (char*)d_ws;
  size_t off = 0;
  auto alloc = [&](size_t bytes) -> char* {
    char* p = w + off;
    off = (off + bytes + 255) & ~(size_t)255;
    return p;
  };

  __bf16* in_b = (__bf16*)alloc((size_t)M * D_IN * 2);
  __bf16* ew_b = (__bf16*)alloc((size_t)D * D_IN * 2);
  __bf16* wqb  = (__bf16*)alloc((size_t)D * D * 2);
  __bf16* wkb  = (__bf16*)alloc((size_t)D * D * 2);
  __bf16* wvb  = (__bf16*)alloc((size_t)D * D * 2);
  __bf16* swb  = (__bf16*)alloc((size_t)D * D * 2);
  __bf16* xb   = (__bf16*)alloc((size_t)M * D * 2);
  float*  xf   = (float*)alloc((size_t)M * D * 4);
  __bf16* Qb   = (__bf16*)alloc((size_t)M * D * 2);
  __bf16* Kb   = (__bf16*)alloc((size_t)M * D * 2);
  __bf16* Vb   = (__bf16*)alloc((size_t)M * D * 2);
  float*  Zf   = (float*)alloc((size_t)M * D * 4);
  float*  Hf   = (float*)alloc((size_t)M * D * 4);
  __bf16* Hb   = (__bf16*)alloc((size_t)M * D * 2);
  float*  SK   = Zf;  // alias: Zf fully consumed by add_layernorm before reuse
  float*  tb   = (float*)alloc((size_t)M * 4);
  float*  pb   = (float*)alloc((size_t)M * 4);

  // 1) fp32 -> bf16 conversions
  f32_to_bf16_k<<<(M * D_IN) / 256, 256, 0, stream>>>(inputs, in_b);
  f32_to_bf16_k<<<(D * D_IN) / 256, 256, 0, stream>>>(embed_w, ew_b);
  f32_to_bf16_k<<<(D * D) / 256, 256, 0, stream>>>(wq_w, wqb);
  f32_to_bf16_k<<<(D * D) / 256, 256, 0, stream>>>(wk_w, wkb);
  f32_to_bf16_k<<<(D * D) / 256, 256, 0, stream>>>(wv_w, wvb);
  f32_to_bf16_k<<<(D * D) / 256, 256, 0, stream>>>(sw_w, swb);

  const dim3 gg(D / 128, M / 128);

  // 2) embed + posenc
  gemm_bf16<<<gg, 256, 0, stream>>>(in_b, ew_b, embed_b, pos, xf, xb, D_IN, 0);

  // 3) Q, K, V projections
  gemm_bf16<<<gg, 256, 0, stream>>>(xb, wqb, wq_b, nullptr, nullptr, Qb, D, 1);
  gemm_bf16<<<gg, 256, 0, stream>>>(xb, wkb, wk_b, nullptr, nullptr, Kb, D, 1);
  gemm_bf16<<<gg, 256, 0, stream>>>(xb, wvb, wv_b, nullptr, nullptr, Vb, D, 1);

  // 4) flash attention -> Zf
  const size_t smem =
      (size_t)(16 * 1024 + 2 * 128 * 32 + 16 * 128 + 32 * 1024) * sizeof(__bf16) +
      320 * sizeof(float);  // ~117 KB -> 2 workgroups per 320 KB WGP
  flash_attn<<<dim3(SEQ / 16, B), 256, smem, stream>>>(Qb, Kb, Vb, Zf);

  // 5) residual + LayerNorm
  add_layernorm<<<M, 256, 0, stream>>>(xf, Zf, ln_g, ln_b, Hf, Hb);

  // 6) soft_key = tanh(H @ soft_key_w^T)
  gemm_bf16<<<gg, 256, 0, stream>>>(Hb, swb, nullptr, nullptr, SK, nullptr, D, 2);

  // 7..9) pooling softmax
  softkey_dot<<<M, 256, 0, stream>>>(SK, sq, tb);
  seq_softmax<<<B, 256, 0, stream>>>(tb, pb);
  weighted_sum<<<dim3(D / 256, B), 256, 0, stream>>>(pb, Hf, out);
}
